// STGN_LSTM_22196390986556
// MI455X (gfx1250) — compile-verified, matrix operand order checked by reading, not run on hardware
//
#include <hip/hip_runtime.h>
#include <math.h>

#define Hdim 256
#define Bsz  512
#define Tlen 100
#define Mmem 20

typedef __attribute__((ext_vector_type(16))) __bf16 v16bf;
typedef __attribute__((ext_vector_type(8)))  float  v8f;

union FragAB { v16bf v; uint4 u[2]; };

__device__ __forceinline__ unsigned short f2bf(float f) {
    unsigned int u = __float_as_uint(f);
    unsigned int r = u + 0x7FFFu + ((u >> 16) & 1u);
    return (unsigned short)(r >> 16);
}
__device__ __forceinline__ float bf2f(unsigned short s) {
    return __uint_as_float(((unsigned int)s) << 16);
}
__device__ __forceinline__ float sigmoidf(float x) {
    return 1.0f / (1.0f + expf(-x));
}

// ---------------------------------------------------------------------------
// Repack fp32 (K=256 x N=256) row-major weights into WMMA-B bf16 fragments.
// Fragment layout for V_WMMA_F32_16X16X32_BF16 B-operand (32x16, K x N):
//   lane l: N = l%16 ; khalf = (l<16)?0:8
//   elems 0..7  -> K = kt*32 + khalf + e
//   elems 8..15 -> K = kt*32 + 16 + khalf + (e-8)
// dst element offset: w*65536 + ((kt*16 + nt)*32 + lane)*16 + e
// ---------------------------------------------------------------------------
struct RepackP { const float* src[9]; };

__global__ __launch_bounds__(32) void repack_kernel(RepackP P, unsigned short* __restrict__ wp) {
    int lane = threadIdx.x;          // 0..31
    int tile = blockIdx.x;           // 0..127 : kt*16 + nt
    int w    = blockIdx.y;           // 0..8
    int kt = tile >> 4, nt = tile & 15;
    int n = nt * 16 + (lane & 15);
    int khalf = (lane >> 4) * 8;
    const float* W = P.src[w];
    unsigned short* dst = wp + (size_t)w * 65536 + ((size_t)(kt * 16 + nt) * 32 + lane) * 16;
#pragma unroll
    for (int e = 0; e < 8; ++e)
        dst[e] = f2bf(W[(kt * 32 + khalf + e) * Hdim + n]);
#pragma unroll
    for (int e = 0; e < 8; ++e)
        dst[8 + e] = f2bf(W[(kt * 32 + 16 + khalf + e) * Hdim + n]);
}

// ---------------------------------------------------------------------------
// Precompute zeta for ALL timesteps (no recurrence):
// zeta[t,b,h] = tanh(dt*WzT_w[h] + WzT_b[h] + dL*WzL_w[h] + WzL_b[h])  (bf16)
// ---------------------------------------------------------------------------
__global__ __launch_bounds__(256) void zeta_all_kernel(
    const float* __restrict__ X,
    const float* __restrict__ WzT_w, const float* __restrict__ WzT_b,
    const float* __restrict__ WzL_w, const float* __restrict__ WzL_b,
    unsigned short* __restrict__ zeta)
{
    size_t idx = (size_t)blockIdx.x * 256 + threadIdx.x;  // t*B*H + b*H + h
    int t = (int)(idx >> 17);              // B*H = 131072 = 2^17
    int rem = (int)(idx & 131071);
    int b = rem >> 8, h = rem & 255;
    float dt = X[(b * Tlen + t) * 4 + 2];
    float dL = X[(b * Tlen + t) * 4 + 3];
    zeta[idx] = f2bf(tanhf(dt * WzT_w[h] + WzT_b[h] + dL * WzL_w[h] + WzL_b[h]));
}

// ---------------------------------------------------------------------------
// Fused gate GEMMs + LSTM cell update.
// Each wave: one 16x16 (batch x H) tile, 7 accumulators (f,i,T,L,u,o,oz).
// A fragment (h_prev / zeta rows) loaded once per K-step, reused across the
// 7 weight matrices.
// ---------------------------------------------------------------------------
struct GateP {
    const float *Wfh_b, *Wih_b, *WTh_b, *WLh_b, *Wuh_b, *Woh_b, *Woz_b;
    const float *Wfx_w, *Wfx_b, *bf;
    const float *Wix_w, *Wix_b, *bi;
    const float *WTx_w, *WTx_b, *WTt_w, *WTt_b, *bT;
    const float *WLx_w, *WLx_b, *WLd_w, *WLd_b, *bL;
    const float *Wux_w, *Wux_b, *bu;
    const float *Wox_w, *Wox_b, *bo;
};

__global__ __launch_bounds__(128) void gates_kernel(
    const unsigned short* __restrict__ hp,    // bf16 B x H  (h_prev == s)
    const unsigned short* __restrict__ zeta,  // bf16 B x H  (this step's slice)
    const unsigned short* __restrict__ wp,    // packed weights (9 mats)
    float* __restrict__ cbuf,                 // f32 B x H   (in-place c)
    unsigned short* __restrict__ bufht,       // bf16: B*M rows (buf) + B rows (h_t)
    const float* __restrict__ X,
    GateP P, int t)
{
    const int lane = threadIdx.x & 31;
    const int wid  = threadIdx.x >> 5;
    const int tile = blockIdx.x * 4 + wid;    // 0..511
    const int mt = tile >> 4, nt = tile & 15;

    v8f zf = {0.f, 0.f, 0.f, 0.f, 0.f, 0.f, 0.f, 0.f};
    v8f acc[7];
#pragma unroll
    for (int g = 0; g < 7; ++g) acc[g] = zf;

    const int m = mt * 16 + (lane & 15);
    const int khalf = (lane >> 4) * 8;
    const uint4* hrow = (const uint4*)(hp   + (size_t)m * Hdim);
    const uint4* zrow = (const uint4*)(zeta + (size_t)m * Hdim);

    for (int kt = 0; kt < 8; ++kt) {
        const int ke = kt * 32 + khalf;       // bf16 element offset in row
        FragAB ah, az;
        ah.u[0] = hrow[ke >> 3];
        ah.u[1] = hrow[(ke + 16) >> 3];
        az.u[0] = zrow[ke >> 3];
        az.u[1] = zrow[(ke + 16) >> 3];
        if (kt < 7) {
            __builtin_prefetch((const void*)((const uint4*)wp +
                ((size_t)((kt + 1) * 16 + nt) * 32 + lane) * 2), 0, 3);
        }
#pragma unroll
        for (int g = 0; g < 7; ++g) {
            const uint4* wg = (const uint4*)(wp + (size_t)g * 65536) +
                              ((size_t)(kt * 16 + nt) * 32 + lane) * 2;
            FragAB bw;
            bw.u[0] = wg[0];
            bw.u[1] = wg[1];
            acc[g] = __builtin_amdgcn_wmma_f32_16x16x32_bf16(
                false, (g == 6) ? az.v : ah.v, false, bw.v,
                (short)0, acc[g], false, false);
        }
    }

    // ---- epilogue: N is constant per lane across the 8 C rows ----
    const int n = nt * 16 + (lane & 15);
    const float bfh = P.Wfh_b[n], bih = P.Wih_b[n], bTh = P.WTh_b[n];
    const float bLh = P.WLh_b[n], buh = P.Wuh_b[n], boh = P.Woh_b[n];
    const float boz = P.Woz_b[n];
    const float wfx0 = P.Wfx_w[n], wfx1 = P.Wfx_w[Hdim + n], bfx = P.Wfx_b[n], bff = P.bf[n];
    const float wix0 = P.Wix_w[n], wix1 = P.Wix_w[Hdim + n], bix = P.Wix_b[n], bii = P.bi[n];
    const float wTx0 = P.WTx_w[n], wTx1 = P.WTx_w[Hdim + n], bTx = P.WTx_b[n];
    const float wTt  = P.WTt_w[n], bTt = P.WTt_b[n], bTT = P.bT[n];
    const float wLx0 = P.WLx_w[n], wLx1 = P.WLx_w[Hdim + n], bLx = P.WLx_b[n];
    const float wLd  = P.WLd_w[n], bLd = P.WLd_b[n], bLL = P.bL[n];
    const float wux0 = P.Wux_w[n], wux1 = P.Wux_w[Hdim + n], bux = P.Wux_b[n], buu = P.bu[n];
    const float wox0 = P.Wox_w[n], wox1 = P.Wox_w[Hdim + n], box = P.Wox_b[n], boo = P.bo[n];

    const int slot = t % Mmem;
    unsigned short* htp = bufht + (size_t)Bsz * Mmem * Hdim;

#pragma unroll
    for (int r = 0; r < 8; ++r) {
        const int b = mt * 16 + (lane >> 4) * 8 + r;
        float4 xv = ((const float4*)X)[b * Tlen + t];
        const float x0 = xv.x, x1 = xv.y, dt = xv.z, dL = xv.w;
        float f  = sigmoidf(acc[0][r] + bfh + x0 * wfx0 + x1 * wfx1 + bfx + bff);
        float i  = sigmoidf(acc[1][r] + bih + x0 * wix0 + x1 * wix1 + bix + bii);
        float Tg = sigmoidf(acc[2][r] + bTh + x0 * wTx0 + x1 * wTx1 + bTx + dt * wTt + bTt + bTT);
        float Lg = sigmoidf(acc[3][r] + bLh + x0 * wLx0 + x1 * wLx1 + bLx + dL * wLd + bLd + bLL);
        float ct = tanhf  (acc[4][r] + buh + x0 * wux0 + x1 * wux1 + bux + buu);
        float o  = sigmoidf(acc[5][r] + boh + x0 * wox0 + x1 * wox1 + box + acc[6][r] + boz + boo);
        float cprev = cbuf[(size_t)b * Hdim + n];
        float cnew  = f * cprev + i * ct * Tg * Lg;
        cbuf[(size_t)b * Hdim + n] = cnew;
        float hv = o * tanhf(cnew);
        unsigned short hb = f2bf(hv);
        htp[(size_t)b * Hdim + n] = hb;
        bufht[((size_t)b * Mmem + slot) * Hdim + n] = hb;
    }
}

// ---------------------------------------------------------------------------
// Combined attention GEMM: rows 0..B*M-1 = buf @ WWh (+WWh_b),
//                          rows B*M..B*M+B-1 = h_t @ WIh (+WIh_b).
// Each wave: 4 consecutive 16-row tiles x one 16-col tile.  The B (weight)
// fragment is loaded once per K-step and reused across the 4 row tiles
// (4x operand reuse, 32 WMMAs per wave).
// ---------------------------------------------------------------------------
__global__ __launch_bounds__(128) void attgemm_kernel(
    const unsigned short* __restrict__ A,   // bufht (rows contiguous)
    const unsigned short* __restrict__ wp,  // packed weights base
    float* __restrict__ out,                // (B*M + B) x H fp32
    const float* __restrict__ WWh_b,
    const float* __restrict__ WIh_b)
{
    const int lane = threadIdx.x & 31;
    const int wid  = threadIdx.x >> 5;
    const int tile = blockIdx.x * 4 + wid;  // 0..(168*16-1)
    const int st = tile >> 4;               // super row-tile: rows st*64..st*64+63
    const int nt = tile & 15;

    // boundary at row-tile 640 == super-tile 160 (buf vs h_t region)
    const unsigned short* wsel = wp + (size_t)((st < 160) ? 7 : 8) * 65536;
    const float* bias = (st < 160) ? WWh_b : WIh_b;

    v8f zf = {0.f, 0.f, 0.f, 0.f, 0.f, 0.f, 0.f, 0.f};
    v8f acc[4];
#pragma unroll
    for (int j = 0; j < 4; ++j) acc[j] = zf;

    const int khalf = (lane >> 4) * 8;
    const uint4* arow[4];
#pragma unroll
    for (int j = 0; j < 4; ++j) {
        const int m = (st * 4 + j) * 16 + (lane & 15);
        arow[j] = (const uint4*)(A + (size_t)m * Hdim);
    }

    for (int kt = 0; kt < 8; ++kt) {
        const int ke = kt * 32 + khalf;
        FragAB bw;
        const uint4* wg = (const uint4*)wsel + ((size_t)(kt * 16 + nt) * 32 + lane) * 2;
        bw.u[0] = wg[0];
        bw.u[1] = wg[1];
        if (kt < 7) {
            __builtin_prefetch((const void*)((const uint4*)wsel +
                ((size_t)((kt + 1) * 16 + nt) * 32 + lane) * 2), 0, 3);
        }
#pragma unroll
        for (int j = 0; j < 4; ++j) {
            FragAB af;
            af.u[0] = arow[j][ke >> 3];
            af.u[1] = arow[j][(ke + 16) >> 3];
            acc[j] = __builtin_amdgcn_wmma_f32_16x16x32_bf16(
                false, af.v, false, bw.v, (short)0, acc[j], false, false);
        }
    }

    const int n = nt * 16 + (lane & 15);
    const float bn = bias[n];
#pragma unroll
    for (int j = 0; j < 4; ++j) {
#pragma unroll
        for (int r = 0; r < 8; ++r) {
            const int row = (st * 4 + j) * 16 + (lane >> 4) * 8 + r;
            out[(size_t)row * Hdim + n] = acc[j][r] + bn;
        }
    }
}

// ---------------------------------------------------------------------------
// Attention finish: one workgroup per batch element.
// o_att = tanh(Wmem + q + dt*WIAT + dL*WIAL + bo_att)
// scores[i,j] = o_att[i] . mem[j]  (masked softmax over j)
// s[h] = (sum_i (sum_j alpha[i,j]) * mem[i,h]) / m_cnt   -> next h_prev (bf16)
// ---------------------------------------------------------------------------
__global__ __launch_bounds__(256) void attfinal_kernel(
    const float* __restrict__ attout,        // Wmem rows then q rows
    const unsigned short* __restrict__ bufht,// buf rows (mem source)
    const float* __restrict__ X,
    const float* __restrict__ WIAT_w, const float* __restrict__ WIAT_b,
    const float* __restrict__ WIAL_w, const float* __restrict__ WIAL_b,
    const float* __restrict__ bo_att,
    unsigned short* __restrict__ hnext, int t)
{
    __shared__ float oatt[Mmem][Hdim];
    __shared__ float memv[Mmem][Hdim];
    __shared__ float alpha[Mmem][Mmem];
    __shared__ float rowsum[Mmem];

    const int b = blockIdx.x;
    const int h = threadIdx.x;
    const int mcnt = (t + 1 < Mmem) ? (t + 1) : Mmem;
    const float dt = X[(b * Tlen + t) * 4 + 2];
    const float dL = X[(b * Tlen + t) * 4 + 3];

    const float* q = attout + ((size_t)Bsz * Mmem + b) * Hdim;
    const float extra = q[h] + dt * WIAT_w[h] + WIAT_b[h]
                             + dL * WIAL_w[h] + WIAL_b[h] + bo_att[h];
    for (int i = 0; i < Mmem; ++i) {
        float wm = attout[((size_t)b * Mmem + i) * Hdim + h];
        oatt[i][h] = tanhf(wm + extra);
        memv[i][h] = (i < mcnt) ? bf2f(bufht[((size_t)b * Mmem + i) * Hdim + h]) : 0.0f;
    }
    __syncthreads();

    for (int p = threadIdx.x; p < Mmem * Mmem; p += 256) {
        int i = p / Mmem, j = p % Mmem;
        float s = 0.0f;
        for (int k = 0; k < Hdim; ++k) s += oatt[i][k] * memv[j][k];
        alpha[i][j] = s;
    }
    __syncthreads();

    if (threadIdx.x < Mmem) {
        const int i = threadIdx.x;
        float mx = -1e30f;
        for (int j = 0; j < mcnt; ++j) mx = fmaxf(mx, alpha[i][j]);
        float e[Mmem];
        float sum = 0.0f;
        for (int j = 0; j < Mmem; ++j) {
            float v = (j < mcnt) ? expf(alpha[i][j] - mx) : 0.0f;
            e[j] = v; sum += v;
        }
        float inv = 1.0f / sum;
        float rs = 0.0f;
        for (int j = 0; j < Mmem; ++j) { float a = e[j] * inv; alpha[i][j] = a; rs += a; }
        rowsum[i] = rs;
    }
    __syncthreads();

    float s = 0.0f;
    for (int i = 0; i < Mmem; ++i) s += rowsum[i] * memv[i][h];
    s /= (float)mcnt;
    hnext[(size_t)b * Hdim + h] = f2bf(s);
}

// ---------------------------------------------------------------------------
// Final: out[b] = s_final[b,:] . fc_w + fc_b
// ---------------------------------------------------------------------------
__global__ __launch_bounds__(256) void fc_kernel(
    const unsigned short* __restrict__ hfin,
    const float* __restrict__ fc_w, const float* __restrict__ fc_b,
    float* __restrict__ out)
{
    int b = blockIdx.x * blockDim.x + threadIdx.x;
    if (b >= Bsz) return;
    float s = 0.0f;
    for (int k = 0; k < Hdim; ++k) s += bf2f(hfin[(size_t)b * Hdim + k]) * fc_w[k];
    out[b] = s + fc_b[0];
}

// ---------------------------------------------------------------------------
extern "C" void kernel_launch(void* const* d_in, const int* in_sizes, int n_in,
                              void* d_out, int out_size, void* d_ws, size_t ws_size,
                              hipStream_t stream)
{
    (void)in_sizes; (void)n_in; (void)out_size; (void)ws_size;
    const float* X = (const float*)d_in[0];
    auto F = [&](int i) { return (const float*)d_in[i]; };

    // workspace carve-up (256B aligned)
    char* ws = (char*)d_ws;
    size_t off = 0;
    auto carve = [&](size_t bytes) { size_t o = off; off += (bytes + 255) & ~(size_t)255; return o; };
    unsigned short* wp    = (unsigned short*)(ws + carve((size_t)9 * 65536 * 2));
    unsigned short* hA    = (unsigned short*)(ws + carve((size_t)Bsz * Hdim * 2));
    unsigned short* hB    = (unsigned short*)(ws + carve((size_t)Bsz * Hdim * 2));
    unsigned short* zetaA = (unsigned short*)(ws + carve((size_t)Tlen * Bsz * Hdim * 2));
    float*          cbuf  = (float*)         (ws + carve((size_t)Bsz * Hdim * 4));
    unsigned short* bufht = (unsigned short*)(ws + carve((size_t)(Bsz * Mmem + Bsz) * Hdim * 2));
    float*          atto  = (float*)         (ws + carve((size_t)(Bsz * Mmem + Bsz) * Hdim * 4));

    // 1) repack the 9 HxH weights into WMMA-B bf16 fragments
    RepackP rp;
    rp.src[0] = F(1);  // Wfh_w
    rp.src[1] = F(5);  // Wih_w
    rp.src[2] = F(9);  // WTh_w
    rp.src[3] = F(15); // WLh_w
    rp.src[4] = F(25); // Wuh_w
    rp.src[5] = F(29); // Woh_w
    rp.src[6] = F(33); // Woz_w
    rp.src[7] = F(37); // WWh_w
    rp.src[8] = F(35); // WIh_w
    repack_kernel<<<dim3(128, 9), 32, 0, stream>>>(rp, wp);

    // 2) precompute zeta for all timesteps (no recurrence)
    zeta_all_kernel<<<(Tlen * Bsz * Hdim) / 256, 256, 0, stream>>>(
        X, F(21), F(22), F(23), F(24), zetaA);

    // 3) zero initial state (h0, c0, buf0)
    hipMemsetAsync(hA,    0, (size_t)Bsz * Hdim * 2, stream);
    hipMemsetAsync(cbuf,  0, (size_t)Bsz * Hdim * 4, stream);
    hipMemsetAsync(bufht, 0, (size_t)Bsz * Mmem * Hdim * 2, stream);

    GateP gp;
    gp.Wfh_b = F(2);  gp.Wih_b = F(6);  gp.WTh_b = F(10); gp.WLh_b = F(16);
    gp.Wuh_b = F(26); gp.Woh_b = F(30); gp.Woz_b = F(34);
    gp.Wfx_w = F(3);  gp.Wfx_b = F(4);  gp.bf = F(45);
    gp.Wix_w = F(7);  gp.Wix_b = F(8);  gp.bi = F(46);
    gp.WTx_w = F(11); gp.WTx_b = F(12); gp.WTt_w = F(13); gp.WTt_b = F(14); gp.bT = F(47);
    gp.WLx_w = F(17); gp.WLx_b = F(18); gp.WLd_w = F(19); gp.WLd_b = F(20); gp.bL = F(48);
    gp.Wux_w = F(27); gp.Wux_b = F(28); gp.bu = F(49);
    gp.Wox_w = F(31); gp.Wox_b = F(32); gp.bo = F(50);

    unsigned short* hp = hA;
    unsigned short* hn = hB;
    for (int t = 0; t < Tlen; ++t) {
        gates_kernel<<<(Bsz / 16) * (Hdim / 16) / 4, 128, 0, stream>>>(
            hp, zetaA + (size_t)t * Bsz * Hdim, wp, cbuf, bufht, X, gp, t);
        attgemm_kernel<<<((Bsz * Mmem + Bsz) / 64) * (Hdim / 16) / 4, 128, 0, stream>>>(
            bufht, wp, atto, F(38), F(36));
        attfinal_kernel<<<Bsz, 256, 0, stream>>>(
            atto, bufht, X, F(39), F(40), F(41), F(42), F(51), hn, t);
        unsigned short* tmp = hp; hp = hn; hn = tmp;
    }

    fc_kernel<<<2, 256, 0, stream>>>(hp, F(43), F(44), (float*)d_out);
}
